// GraphEncoder_74723841016374
// MI455X (gfx1250) — matrix operand order repaired and checked by reading, not compiled, not run on hardware
//
#include <hip/hip_runtime.h>
#include <hip/hip_bf16.h>
#include <math.h>

// ---------------------------------------------------------------------------
// GraphEncoder on MI455X (gfx1250, wave32, WMMA).
//
// Stage 1 (dominant, ~137 GFLOP): f[n,k] = (1/B) * sum_b relu( relu(x[b,n]*w1 + b1) @ W2 + b2 )
//   -> cast as a 1M x 256 x 256 GEMM with A generated on the fly from x,
//      computed with v_wmma_f32_16x16x32_f16, with the mean-over-b reduction
//      kept in the f32 accumulator registers (GEMM output never materialized).
//      x traffic is only 4 MB (0.2 us at 23.3 TB/s) -> stage is matrix-core bound.
// Stage 2 (tiny): A' = f@W3[:H] + b3 ; Bp = f@W3[H:]
// Stage 3 (~200M VALU): s[i,j] = sigmoid( sum_k relu(A'[i,k]+Bp[j,k]) * W4[k] + b4 ), diag=0
// ---------------------------------------------------------------------------

typedef _Float16 v16h __attribute__((ext_vector_type(16)));
typedef _Float16 v8h  __attribute__((ext_vector_type(8)));
typedef float    v8f  __attribute__((ext_vector_type(8)));

#define HDIM   256
#define NTOT   512
#define BTOT   2048
#define BCHUNK 64     // b-values per block
#define HPAD   8      // fp16 pad so row stride = 528B (16B aligned, bank-spread)

// ---------------------------------------------------------------------------
// Kernel 0: zero the f accumulator; build W2h[n][k] = (f16) W2[k][n]
// ---------------------------------------------------------------------------
__global__ __launch_bounds__(256) void ge_prep_kernel(
    const float* __restrict__ W2, _Float16* __restrict__ W2h,
    float* __restrict__ f_acc)
{
    int idx = blockIdx.x * 256 + threadIdx.x;
    if (idx < NTOT * HDIM) f_acc[idx] = 0.0f;
    if (idx < HDIM * HDIM) {
        int n = idx >> 8, k = idx & 255;
        W2h[n * HDIM + k] = (_Float16)W2[k * HDIM + n];
    }
}

// ---------------------------------------------------------------------------
// Kernel 1: the WMMA GEMM + mean reduction into f.
// Block = 512 threads = 16 waves. Block owns (16 n-values) x (64 b-values).
// Wave w owns output column-tile w (16 of the 256 H columns); its B-panel
// (W2 K=0..255 x its 16 columns, fp16) lives in 8 x v16h registers.
// Per b (row-tile): build the 16x256 fp16 h-panel in LDS (double-buffered,
// one barrier per iteration), preload all 8 A tiles in one ds burst, then
// run 8 back-to-back WMMAs and accumulate relu(c+b2) into F registers.
// ---------------------------------------------------------------------------
__global__ __launch_bounds__(512) void ge_fenc_kernel(
    const float* __restrict__ x,
    const float* __restrict__ W1, const float* __restrict__ b1,
    const _Float16* __restrict__ W2h, const float* __restrict__ b2,
    float* __restrict__ f_acc)
{
    __shared__ float    xs[BCHUNK][16];
    __shared__ _Float16 hs[2][16][HDIM + HPAD];

    const int tid = threadIdx.x;
    const int nb  = blockIdx.x & 31;   // 32 n-tiles
    const int bc  = blockIdx.x >> 5;   // 32 b-chunks
    const int n0  = nb * 16;
    const int b0  = bc * BCHUNK;

    // ---- wave/lane identity (wave == column tile)
    const int wave = tid >> 5;      // 0..15
    const int lane = tid & 31;
    const int ln   = lane & 15;     // N index within tile / A row
    const int lh   = lane >> 4;     // K-half selector
    const int col  = wave * 16 + ln;

    __builtin_prefetch(W2h + col * HDIM, 0, 3);   // global_prefetch_b8 on B panel

    // ---- stage x slab: x[b0..b0+63, n0..n0+15] -> LDS
    for (int idx = tid; idx < BCHUNK * 16; idx += 512) {
        int t = idx >> 4, m = idx & 15;
        xs[t][m] = x[(b0 + t) * NTOT + n0 + m];
    }

    // ---- per-thread h-build constants: row hm, 8 consecutive k's
    const int hm = tid >> 5;        // 0..15
    const int hk = (tid & 31) * 8;  // 0..248
    float w1v[8], b1v[8];
#pragma unroll
    for (int i = 0; i < 8; ++i) { w1v[i] = W1[hk + i]; b1v[i] = b1[hk + i]; }

    // ---- B panel in registers: lane ln holds column `col`, K split per ISA
    v16h breg[8];
#pragma unroll
    for (int s = 0; s < 8; ++s) {
        const _Float16* p = W2h + col * HDIM + s * 32 + lh * 8;
        v8h lo = *(const v8h*)p;
        v8h hi = *(const v8h*)(p + 16);
        breg[s] = __builtin_shufflevector(lo, hi,
                   0,1,2,3,4,5,6,7,8,9,10,11,12,13,14,15);
    }
    const float b2c = b2[col];

    v8f F = {0.f,0.f,0.f,0.f,0.f,0.f,0.f,0.f};   // sum_b relu(...) for this lane's 8 outputs

    __syncthreads();   // xs staged

    for (int t = 0; t < BCHUNK; ++t) {
        const int buf = t & 1;

        // build h-panel into buffer `buf`: hs[buf][m][k] = f16(relu(x*w1[k]+b1[k]))
        {
            const float xv = xs[t][hm];
            v8h hv;
#pragma unroll
            for (int i = 0; i < 8; ++i) {
                float v = fmaf(xv, w1v[i], b1v[i]);
                hv[i] = (_Float16)(v > 0.f ? v : 0.f);
            }
            *(v8h*)&hs[buf][hm][hk] = hv;        // single ds_store_b128
        }
        __syncthreads();   // h-panel `buf` ready (also fences reuse of other buffer)

        // preload all 8 A tiles as one ds_load burst (ISA 16-bit A 16x32 layout:
        // lane (lh,ln): elems 0-7 -> K=32s+8*lh+i ; elems 8-15 -> K=32s+16+8*lh+i)
        v16h a[8];
#pragma unroll
        for (int s = 0; s < 8; ++s) {
            const _Float16* ap = &hs[buf][ln][s * 32 + lh * 8];
            v8h a0 = *(const v8h*)ap;
            v8h a1 = *(const v8h*)(ap + 16);
            a[s] = __builtin_shufflevector(a0, a1,
                      0,1,2,3,4,5,6,7,8,9,10,11,12,13,14,15);
        }

        // 8 back-to-back WMMAs, K = 256
        v8f c = {0.f,0.f,0.f,0.f,0.f,0.f,0.f,0.f};
#pragma unroll
        for (int s = 0; s < 8; ++s) {
            c = __builtin_amdgcn_wmma_f32_16x16x32_f16(
                    false, a[s], false, breg[s], (short)0, c, false, false);
        }

        // g = relu(c + b2); accumulate the mean-over-b in registers
#pragma unroll
        for (int v = 0; v < 8; ++v) {
            float g = c[v] + b2c;
            F[v] += (g > 0.f ? g : 0.f);
        }
    }

    // ---- fold this block's partial sum into f (f32 atomics, scaled by 1/B)
    const float scale = 1.0f / (float)BTOT;
#pragma unroll
    for (int v = 0; v < 8; ++v) {
        int m = lh * 8 + v;                 // C layout: M = (lane>>4)*8 + v
        atomicAdd(&f_acc[(n0 + m) * HDIM + col], F[v] * scale);
    }
}

// ---------------------------------------------------------------------------
// Kernel 2: A'[n,c] = f[n,:]@W3[:H,c] + b3[c] ;  Bp[n,c] = f[n,:]@W3[H:,c]
// One block per (n, half); f row staged in LDS; W3 reads coalesced over c.
// ---------------------------------------------------------------------------
__global__ __launch_bounds__(256) void ge_ab_kernel(
    const float* __restrict__ f, const float* __restrict__ W3,
    const float* __restrict__ b3,
    float* __restrict__ Ap, float* __restrict__ Bp)
{
    __shared__ float fs[HDIM];
    const int n    = blockIdx.x >> 1;
    const int half = blockIdx.x & 1;
    const int c    = threadIdx.x;

    fs[c] = f[n * HDIM + c];
    __syncthreads();

    const float* W3p = W3 + half * HDIM * HDIM;
    float acc = 0.f;
#pragma unroll 4
    for (int k = 0; k < HDIM; ++k)
        acc = fmaf(fs[k], W3p[k * HDIM + c], acc);

    if (half == 0) Ap[n * HDIM + c] = acc + b3[c];   // fold b3 into A'
    else           Bp[n * HDIM + c] = acc;
}

// ---------------------------------------------------------------------------
// Kernel 3: s[i,j] = sigmoid( sum_k relu(A'[i,k]+Bp[j,k]) * W4[k] + b4 ); diag 0
// Block = 16x16 (i,j) tile; A'/Bp rows + W4 staged in LDS (rows padded).
// ---------------------------------------------------------------------------
__global__ __launch_bounds__(256) void ge_edge_kernel(
    const float* __restrict__ Ap, const float* __restrict__ Bp,
    const float* __restrict__ W4, const float* __restrict__ b4,
    float* __restrict__ out)
{
    __shared__ float As[16][HDIM + 4];
    __shared__ float Bs[16][HDIM + 4];
    __shared__ float w4s[HDIM];

    const int ib = blockIdx.x & 31, jb = blockIdx.x >> 5;
    const int i0 = ib * 16, j0 = jb * 16;
    const int tid = threadIdx.x;

    for (int idx = tid; idx < 16 * HDIM; idx += 256) {
        int r = idx >> 8, k = idx & 255;
        As[r][k] = Ap[(i0 + r) * HDIM + k];
        Bs[r][k] = Bp[(j0 + r) * HDIM + k];
    }
    w4s[tid] = W4[tid];
    __syncthreads();

    const int ti = tid >> 4, tj = tid & 15;
    float acc = 0.f;
#pragma unroll 4
    for (int k = 0; k < HDIM; ++k) {
        float e = As[ti][k] + Bs[tj][k];
        acc = fmaf(e > 0.f ? e : 0.f, w4s[k], acc);
    }
    acc += b4[0];
    float s = 1.0f / (1.0f + expf(-acc));
    const int i = i0 + ti, j = j0 + tj;
    out[i * NTOT + j] = (i == j) ? 0.f : s;
}

// ---------------------------------------------------------------------------
// Host launcher. Inputs: x, W1, b1, W2, b2, W3, b3, W4, b4 (all f32).
// Workspace layout: [f 512KB][W2h 128KB][A' 512KB][Bp 512KB]
// ---------------------------------------------------------------------------
extern "C" void kernel_launch(void* const* d_in, const int* in_sizes, int n_in,
                              void* d_out, int out_size, void* d_ws, size_t ws_size,
                              hipStream_t stream) {
    const float* x  = (const float*)d_in[0];
    const float* W1 = (const float*)d_in[1];
    const float* b1 = (const float*)d_in[2];
    const float* W2 = (const float*)d_in[3];
    const float* b2 = (const float*)d_in[4];
    const float* W3 = (const float*)d_in[5];
    const float* b3 = (const float*)d_in[6];
    const float* W4 = (const float*)d_in[7];
    const float* b4 = (const float*)d_in[8];

    char* ws = (char*)d_ws;
    float*     f_acc = (float*)(ws);                         // 512 KB
    _Float16*  W2h   = (_Float16*)(ws + 512 * 1024);         // 128 KB
    float*     Ap    = (float*)(ws + 640 * 1024);            // 512 KB
    float*     Bp    = (float*)(ws + 1152 * 1024);           // 512 KB

    // 0) zero f accumulator + transpose/convert W2 to fp16
    ge_prep_kernel<<<(NTOT * HDIM + 255) / 256, 256, 0, stream>>>(W2, W2h, f_acc);

    // 1) WMMA GEMM + mean-over-b: 32 n-tiles x 32 b-chunks
    ge_fenc_kernel<<<32 * 32, 512, 0, stream>>>(x, W1, b1, W2h, b2, f_acc);

    // 2) A' / Bp
    ge_ab_kernel<<<NTOT * 2, 256, 0, stream>>>(f_acc, W3, b3, Ap, Bp);

    // 3) edge scores
    ge_edge_kernel<<<32 * 32, 256, 0, stream>>>(Ap, Bp, W4, b4, (float*)d_out);
}
